// HyperGAT_84473416777730
// MI455X (gfx1250) — compile-verified, matrix operand order checked by reading, not run on hardware
//
#include <hip/hip_runtime.h>
#include <stdint.h>

// ---------------- constants (match reference) ----------------
#define D_IN      256
#define DIM       32
#define HEADS     8
#define NCLS      16
#define MHEDGE    10000

typedef __attribute__((ext_vector_type(16))) _Float16 v16h;
typedef __attribute__((ext_vector_type(8)))  float    v8f;

// ---------------- small helpers ----------------
__device__ __forceinline__ unsigned enc_f32(float f) {
  unsigned u = __float_as_uint(f);
  return (u & 0x80000000u) ? ~u : (u | 0x80000000u);
}
__device__ __forceinline__ float dec_f32(unsigned u) {
  return (u & 0x80000000u) ? __uint_as_float(u ^ 0x80000000u) : __uint_as_float(~u);
}

__global__ void fill0_kernel(float* p, int64_t n) {
  int64_t i = (int64_t)blockIdx.x * blockDim.x + threadIdx.x;
  int64_t stride = (int64_t)gridDim.x * blockDim.x;
  for (; i < n; i += stride) p[i] = 0.0f;
}

// ---------------- degree kernels ----------------
__global__ void degrees_kernel(const int* __restrict__ ni, const int* __restrict__ hi,
                               float* dN, float* dH, int E) {
  int e = blockIdx.x * blockDim.x + threadIdx.x;
  if (e >= E) return;
  atomicAdd(&dN[ni[e]], 1.0f);
  atomicAdd(&dH[hi[e]], 1.0f);
}
__global__ void invert_kernel(float* p, int n) {
  int i = blockIdx.x * blockDim.x + threadIdx.x;
  if (i >= n) return;
  float v = p[i];
  p[i] = (v > 0.0f) ? (1.0f / v) : 0.0f;
}

// ---------------- scatter node features -> hyperedge buckets ----------------
// one thread per (edge, group of 4 channels), C multiple of 4
__global__ void scatter_feat_kernel(const float* __restrict__ src, const int* __restrict__ ni,
                                    const int* __restrict__ hi, float* dst, int E, int C) {
  int g4 = C >> 2;
  int64_t t = (int64_t)blockIdx.x * blockDim.x + threadIdx.x;
  if (t >= (int64_t)E * g4) return;
  int e = (int)(t / g4), g = (int)(t % g4);
  const float4 v = *(const float4*)(src + (size_t)ni[e] * C + g * 4);
  float* d = dst + (size_t)hi[e] * C + g * 4;
  atomicAdd(d + 0, v.x); atomicAdd(d + 1, v.y);
  atomicAdd(d + 2, v.z); atomicAdd(d + 3, v.w);
}

// ---------------- WMMA GEMM: C[M,N] = A[M,K] * B[K,N] (f32 in/out, f16 MACs) ----
// KV: compile-time K (multiple of 32). NT: 16-wide N-subtiles per wave.
// One wave computes a 16 x (16*NT) output tile; M%16==0, N%(16*NT)==0.
// Wave mapping: consecutive waves in a block share the same N-group (B fragments
// identical -> WGP$/L2 broadcast), and walk adjacent M-tiles (A rows coalesce).
template <int KV, int NT>
__global__ void gemm_wmma_f16_kernel(const float* __restrict__ A, const float* __restrict__ B,
                                     float* __restrict__ C, int M, int N) {
  int wavesPerBlock = blockDim.x >> 5;
  int waveId = blockIdx.x * wavesPerBlock + (threadIdx.x >> 5);
  int lane = threadIdx.x & 31;
  int tilesM = M >> 4;
  int groupsN = (N >> 4) / NT;
  int tileM  = waveId % tilesM;
  int groupN = waveId / tilesM;
  if (groupN >= groupsN) return;

  int half = lane >> 4;      // 0: lanes 0-15, 1: lanes 16-31
  int l16  = lane & 15;
  const float* Arow = A + (size_t)(tileM * 16 + l16) * KV;
  int colBase = groupN * NT * 16 + l16;

  v8f acc[NT];
#pragma unroll
  for (int nt = 0; nt < NT; ++nt) acc[nt] = (v8f){};

#pragma unroll
  for (int k0 = 0; k0 < KV; k0 += 32) {
    int kb = k0 + half * 8;
    // A fragment: K = kb..kb+7 (VGPR0-3) and kb+16..kb+23 (VGPR4-7)
    const float4* ap = (const float4*)(Arow + kb);
    float4 a0 = ap[0], a1 = ap[1];
    const float4* ap2 = (const float4*)(Arow + kb + 16);
    float4 a2 = ap2[0], a3 = ap2[1];
    v16h af;
    af[0] = (_Float16)a0.x; af[1] = (_Float16)a0.y; af[2] = (_Float16)a0.z; af[3] = (_Float16)a0.w;
    af[4] = (_Float16)a1.x; af[5] = (_Float16)a1.y; af[6] = (_Float16)a1.z; af[7] = (_Float16)a1.w;
    af[8]  = (_Float16)a2.x; af[9]  = (_Float16)a2.y; af[10] = (_Float16)a2.z; af[11] = (_Float16)a2.w;
    af[12] = (_Float16)a3.x; af[13] = (_Float16)a3.y; af[14] = (_Float16)a3.z; af[15] = (_Float16)a3.w;
#pragma unroll
    for (int nt = 0; nt < NT; ++nt) {
      int colB = colBase + nt * 16;
      // B fragment (K x N row-major): column colB, same K pattern as A
      v16h bf;
#pragma unroll
      for (int j = 0; j < 8; ++j) {
        bf[j]     = (_Float16)B[(size_t)(kb + j) * N + colB];
        bf[j + 8] = (_Float16)B[(size_t)(kb + 16 + j) * N + colB];
      }
      acc[nt] = __builtin_amdgcn_wmma_f32_16x16x32_f16(false, af, false, bf,
                                                       (short)0, acc[nt], false, false);
    }
  }
  // store: lane<16 -> rows 0..7 of tile, lane>=16 -> rows 8..15
  int rowBase = tileM * 16 + half * 8;
#pragma unroll
  for (int nt = 0; nt < NT; ++nt) {
    int colB = colBase + nt * 16;
#pragma unroll
    for (int v = 0; v < 8; ++v)
      C[(size_t)(rowBase + v) * N + colB] = acc[nt][v];
  }
}

// ---------------- attention logits + leaky relu + segment max -----------------
// one thread per (edge, head); att layout [H, 2C]
__global__ void alpha_raw_kernel(const float* __restrict__ xl, const float* __restrict__ hl,
                                 const int* __restrict__ ni, const int* __restrict__ hi,
                                 const float* __restrict__ att, float* alpha,
                                 unsigned* segmax, int E, int H, int C) {
  int64_t t = (int64_t)blockIdx.x * blockDim.x + threadIdx.x;
  if (t >= (int64_t)E * H) return;
  int e = (int)(t / H), h = (int)(t % H);
  int HC = H * C;
  const float* xi = xl + (size_t)ni[e] * HC + h * C;
  const float* xe = hl + (size_t)hi[e] * HC + h * C;
  const float* a1 = att + (size_t)h * 2 * C;
  const float* a2 = a1 + C;
  float s = 0.0f;
  for (int c = 0; c < C; ++c) s += xi[c] * a1[c] + xe[c] * a2[c];
  s = (s < 0.0f) ? 0.2f * s : s;          // leaky_relu(0.2)
  alpha[t] = s;
  atomicMax(segmax + (size_t)hi[e] * H + h, enc_f32(s));
}

// exp(logit - groupmax), accumulate group sum; alpha updated in place to exp value
__global__ void exp_sum_kernel(const int* __restrict__ hi, float* alpha,
                               const unsigned* __restrict__ segmax, float* segsum,
                               int E, int H) {
  int64_t t = (int64_t)blockIdx.x * blockDim.x + threadIdx.x;
  if (t >= (int64_t)E * H) return;
  int e = (int)(t / H), h = (int)(t % H);
  float m = dec_f32(segmax[(size_t)hi[e] * H + h]);
  float ev = __expf(alpha[t] - m);
  alpha[t] = ev;
  atomicAdd(&segsum[(size_t)hi[e] * H + h], ev);
}

// ---------------- message: node -> hyperedge (weighted scatter) ---------------
// one thread per (edge, group of 4 channels of H*C)
__global__ void msg_n2e_kernel(const float* __restrict__ xl, const int* __restrict__ ni,
                               const int* __restrict__ hi, const float* __restrict__ alpha,
                               const float* __restrict__ segsum, const float* __restrict__ Binv,
                               float* edge_out, int E, int H, int C) {
  int HC = H * C, g4 = HC >> 2;
  int64_t t = (int64_t)blockIdx.x * blockDim.x + threadIdx.x;
  if (t >= (int64_t)E * g4) return;
  int e = (int)(t / g4), g = (int)(t % g4);
  int h = (g * 4) / C;
  int he = hi[e];
  float an = alpha[(size_t)e * H + h] / (segsum[(size_t)he * H + h] + 1e-16f);
  float scale = Binv[he] * an;
  const float4 v = *(const float4*)(xl + (size_t)ni[e] * HC + g * 4);
  float* d = edge_out + (size_t)he * HC + g * 4;
  atomicAdd(d + 0, v.x * scale); atomicAdd(d + 1, v.y * scale);
  atomicAdd(d + 2, v.z * scale); atomicAdd(d + 3, v.w * scale);
}

// ---------------- message: hyperedge -> node ---------------------------------
__global__ void msg_e2n_kernel(const float* __restrict__ edge_out, const int* __restrict__ ni,
                               const int* __restrict__ hi, const float* __restrict__ alpha,
                               const float* __restrict__ segsum, const float* __restrict__ Dinv,
                               float* out, int E, int H, int C) {
  int HC = H * C, g4 = HC >> 2;
  int64_t t = (int64_t)blockIdx.x * blockDim.x + threadIdx.x;
  if (t >= (int64_t)E * g4) return;
  int e = (int)(t / g4), g = (int)(t % g4);
  int h = (g * 4) / C;
  int he = hi[e], n = ni[e];
  float an = alpha[(size_t)e * H + h] / (segsum[(size_t)he * H + h] + 1e-16f);
  float scale = Dinv[n] * an;
  const float4 v = *(const float4*)(edge_out + (size_t)he * HC + g * 4);
  float* d = out + (size_t)n * HC + g * 4;
  atomicAdd(d + 0, v.x * scale); atomicAdd(d + 1, v.y * scale);
  atomicAdd(d + 2, v.z * scale); atomicAdd(d + 3, v.w * scale);
}

// ---------------- bias + relu -------------------------------------------------
__global__ void bias_relu_kernel(float* h, const float* __restrict__ b, int64_t n, int C) {
  int64_t i = (int64_t)blockIdx.x * blockDim.x + threadIdx.x;
  int64_t stride = (int64_t)gridDim.x * blockDim.x;
  for (; i < n; i += stride) {
    int c = (int)(i % C);
    float v = h[i] + b[c];
    h[i] = v > 0.0f ? v : 0.0f;
  }
}

// ---------------- bias + log_softmax over 16 classes --------------------------
__global__ void bias_logsoftmax16_kernel(float* out, const float* __restrict__ b, int N) {
  int n = blockIdx.x * blockDim.x + threadIdx.x;
  if (n >= N) return;
  float v[NCLS];
  float m = -3.402823466e38f;
#pragma unroll
  for (int c = 0; c < NCLS; ++c) {
    v[c] = out[(size_t)n * NCLS + c] + b[c];
    m = fmaxf(m, v[c]);
  }
  float s = 0.0f;
#pragma unroll
  for (int c = 0; c < NCLS; ++c) s += __expf(v[c] - m);
  float l = m + __logf(s);
#pragma unroll
  for (int c = 0; c < NCLS; ++c) out[(size_t)n * NCLS + c] = v[c] - l;
}

// ---------------- host-side launcher -----------------------------------------
static inline int cdiv(int64_t a, int64_t b) { return (int)((a + b - 1) / b); }
static inline int capgrid(int64_t n) { int g = cdiv(n, 256); return g > 65535 ? 65535 : (g < 1 ? 1 : g); }

extern "C" void kernel_launch(void* const* d_in, const int* in_sizes, int n_in,
                              void* d_out, int out_size, void* d_ws, size_t ws_size,
                              hipStream_t stream) {
  (void)n_in; (void)out_size; (void)ws_size;
  const float* x    = (const float*)d_in[0];
  const int*   ni   = (const int*)d_in[1];
  const int*   hi   = (const int*)d_in[2];
  const float* W1   = (const float*)d_in[3];
  const float* att1 = (const float*)d_in[4];
  const float* b1   = (const float*)d_in[5];
  const float* W2   = (const float*)d_in[6];
  const float* att2 = (const float*)d_in[7];
  const float* b2   = (const float*)d_in[8];
  float* out = (float*)d_out;

  const int NN = in_sizes[0] / D_IN;   // 50000
  const int E  = in_sizes[1];          // 400000
  const int MH = MHEDGE;               // 10000

  // workspace bump allocation (bytes)
  char* ws = (char*)d_ws;
  float*    xl1    = (float*)(ws + 0);                        // [NN,256]
  float*    hl1    = (float*)(ws + 51200000);                 // [MH,256]
  float*    hattr  = (float*)(ws + 61440000);                 // [MH,256]
  float*    alpha  = (float*)(ws + 71680000);                 // [E,8] (conv2 uses [E,1])
  unsigned* segmax = (unsigned*)(ws + 84480000);              // [MH,8]
  float*    segsum = (float*)(ws + 84800000);                 // [MH,8]
  float*    Dinv   = (float*)(ws + 85120000);                 // [NN]
  float*    Binv   = (float*)(ws + 85320000);                 // [MH]
  float*    eout   = (float*)(ws + 85360000);                 // [MH,256]
  float*    hbuf   = (float*)(ws + 95600000);                 // [NN,256]
  float*    xl2    = (float*)(ws + 146800000);                // [NN,16]
  float*    hl2    = (float*)(ws + 150000000);                // [MH,16]

  const int B = 256;
  const int WPB = B / 32;  // waves per block

  // ---- phase 0: zero accumulators, compute inverse degrees ----
  fill0_kernel<<<capgrid(NN), B, 0, stream>>>(Dinv, NN);
  fill0_kernel<<<capgrid(MH), B, 0, stream>>>(Binv, MH);
  fill0_kernel<<<capgrid((int64_t)MH*256), B, 0, stream>>>(hattr, (int64_t)MH*256);
  fill0_kernel<<<capgrid((int64_t)MH*8), B, 0, stream>>>((float*)segmax, (int64_t)MH*8);
  fill0_kernel<<<capgrid((int64_t)MH*8), B, 0, stream>>>(segsum, (int64_t)MH*8);
  fill0_kernel<<<capgrid((int64_t)MH*256), B, 0, stream>>>(eout, (int64_t)MH*256);
  fill0_kernel<<<capgrid((int64_t)NN*256), B, 0, stream>>>(hbuf, (int64_t)NN*256);
  fill0_kernel<<<capgrid((int64_t)NN*NCLS), B, 0, stream>>>(out, (int64_t)NN*NCLS);

  degrees_kernel<<<cdiv(E, B), B, 0, stream>>>(ni, hi, Dinv, Binv, E);
  invert_kernel<<<cdiv(NN, B), B, 0, stream>>>(Dinv, NN);
  invert_kernel<<<cdiv(MH, B), B, 0, stream>>>(Binv, MH);

  // ---- conv1 (H=8, C=32) ----
  scatter_feat_kernel<<<cdiv((int64_t)E*64, B), B, 0, stream>>>(x, ni, hi, hattr, E, D_IN);

  { int waves = (NN/16) * ((D_IN/16)/4);   // NT=4 -> 16x64 tile per wave
    gemm_wmma_f16_kernel<256,4><<<cdiv(waves, WPB), B, 0, stream>>>(x, W1, xl1, NN, D_IN); }
  { int waves = (MH/16) * ((D_IN/16)/4);
    gemm_wmma_f16_kernel<256,4><<<cdiv(waves, WPB), B, 0, stream>>>(hattr, W1, hl1, MH, D_IN); }

  alpha_raw_kernel<<<cdiv((int64_t)E*HEADS, B), B, 0, stream>>>(xl1, hl1, ni, hi, att1,
                                                                alpha, segmax, E, HEADS, DIM);
  exp_sum_kernel<<<cdiv((int64_t)E*HEADS, B), B, 0, stream>>>(hi, alpha, segmax, segsum, E, HEADS);
  msg_n2e_kernel<<<cdiv((int64_t)E*64, B), B, 0, stream>>>(xl1, ni, hi, alpha, segsum, Binv,
                                                           eout, E, HEADS, DIM);
  msg_e2n_kernel<<<cdiv((int64_t)E*64, B), B, 0, stream>>>(eout, ni, hi, alpha, segsum, Dinv,
                                                           hbuf, E, HEADS, DIM);
  bias_relu_kernel<<<capgrid((int64_t)NN*256), B, 0, stream>>>(hbuf, b1, (int64_t)NN*256, D_IN);

  // ---- conv2 (H=1, C=16) ----
  fill0_kernel<<<capgrid((int64_t)MH*256), B, 0, stream>>>(hattr, (int64_t)MH*256);
  fill0_kernel<<<capgrid(MH), B, 0, stream>>>((float*)segmax, MH);
  fill0_kernel<<<capgrid(MH), B, 0, stream>>>(segsum, MH);
  fill0_kernel<<<capgrid((int64_t)MH*NCLS), B, 0, stream>>>(eout, (int64_t)MH*NCLS);

  scatter_feat_kernel<<<cdiv((int64_t)E*64, B), B, 0, stream>>>(hbuf, ni, hi, hattr, E, D_IN);

  { int waves = (NN/16) * (NCLS/16);       // NT=1 -> 16x16 tile per wave
    gemm_wmma_f16_kernel<256,1><<<cdiv(waves, WPB), B, 0, stream>>>(hbuf, W2, xl2, NN, NCLS); }
  { int waves = (MH/16) * (NCLS/16);
    gemm_wmma_f16_kernel<256,1><<<cdiv(waves, WPB), B, 0, stream>>>(hattr, W2, hl2, MH, NCLS); }

  alpha_raw_kernel<<<cdiv((int64_t)E, B), B, 0, stream>>>(xl2, hl2, ni, hi, att2,
                                                          alpha, segmax, E, 1, NCLS);
  exp_sum_kernel<<<cdiv((int64_t)E, B), B, 0, stream>>>(hi, alpha, segmax, segsum, E, 1);
  msg_n2e_kernel<<<cdiv((int64_t)E*4, B), B, 0, stream>>>(xl2, ni, hi, alpha, segsum, Binv,
                                                          eout, E, 1, NCLS);
  msg_e2n_kernel<<<cdiv((int64_t)E*4, B), B, 0, stream>>>(eout, ni, hi, alpha, segsum, Dinv,
                                                          out, E, 1, NCLS);
  bias_logsoftmax16_kernel<<<cdiv(NN, B), B, 0, stream>>>(out, b2, NN);
}
